// GCNLSTMRawPluginGenderHanded_90692529422803
// MI455X (gfx1250) — matrix operand order repaired and checked by reading, not compiled, not run on hardware
//
#include <hip/hip_runtime.h>
#include <math.h>

// ---------------------------------------------------------------- types
typedef __attribute__((ext_vector_type(16))) __bf16 v16bf;
typedef __attribute__((ext_vector_type(8)))  float  v8f;
typedef int v4i_vs __attribute__((vector_size(16)));   // matches builtin param type

#define H_DIM  640
#define H4     2560
#define NNODES 1024
#define NEDGES 16384
#define BSZ    16
#define LEN_IN 8500

__device__ __forceinline__ unsigned short f2bf(float f) {
  union { float f; unsigned u; } v; v.f = f;
  unsigned r = v.u + 0x7FFFu + ((v.u >> 16) & 1u);   // round-to-nearest-even
  return (unsigned short)(r >> 16);
}

// async global(16B) -> LDS copy, tracked on ASYNCcnt
__device__ __forceinline__ void async_cp16(const unsigned* g, const unsigned* l) {
  __builtin_amdgcn_global_load_async_to_lds_b128((v4i_vs*)g, (v4i_vs*)l, 0, 0);
}

// ------------------------------------------------- pack A: [M,K] f32 -> bf16 WMMA-A tiles
// tile(mt,kt), lane l: row m = mt*16 + (l&15); hi = l>>4
// dword d holds K-pair: k_local = (e<8 ? e : e+8) + hi*8, e = 2d, 2d+1
// ktMajor=1: tile index = kt*Mt + mt (contiguous A k-slices for block staging)
__global__ void pack_a_kernel(const float* __restrict__ A, unsigned* __restrict__ pA,
                              int M, int K, int ldA, int Mt, int Kt, int ktMajor) {
  int gid = blockIdx.x * blockDim.x + threadIdx.x;
  int total = Mt * Kt * 32;
  if (gid >= total) return;
  int lane = gid & 31;
  int tile = gid >> 5;
  int kt = tile % Kt, mt = tile / Kt;
  int m  = mt * 16 + (lane & 15);
  int hi = lane >> 4;
  int tileIdx = ktMajor ? (kt * Mt + mt) : tile;
  unsigned* dst = pA + ((size_t)tileIdx * 32 + lane) * 8;
#pragma unroll
  for (int d = 0; d < 8; ++d) {
    int e0 = 2 * d;
    int k0 = kt * 32 + ((e0 < 8) ? e0 : e0 + 8) + hi * 8;
    int k1 = k0 + 1;
    float f0 = (m < M && k0 < K) ? A[(size_t)m * ldA + k0] : 0.f;
    float f1 = (m < M && k1 < K) ? A[(size_t)m * ldA + k1] : 0.f;
    dst[d] = (unsigned)f2bf(f0) | ((unsigned)f2bf(f1) << 16);
  }
}

// ------------------------------------------------- pack B: logical B[k][n] -> bf16 WMMA-B tiles
// B[k][n] = transB ? W[n*ldw + k] : W[k*ldw + n]; tile index = kt*Nt + nt (kt-major)
__global__ void pack_b_kernel(const float* __restrict__ W, unsigned* __restrict__ pB,
                              int K, int N, int Kt, int Nt, int ldw, int transB) {
  int gid = blockIdx.x * blockDim.x + threadIdx.x;
  int total = Kt * Nt * 32;
  if (gid >= total) return;
  int lane = gid & 31;
  int tile = gid >> 5;
  int nt = tile % Nt, kt = tile / Nt;
  int n  = nt * 16 + (lane & 15);
  int hi = lane >> 4;
  unsigned* dst = pB + ((size_t)tile * 32 + lane) * 8;
#pragma unroll
  for (int d = 0; d < 8; ++d) {
    int k0 = kt * 32 + ((2 * d < 8) ? 2 * d : 2 * d + 8) + hi * 8;
    int k1 = k0 + 1;
    float f0 = 0.f, f1 = 0.f;
    if (n < N) {
      if (k0 < K) f0 = transB ? W[(size_t)n * ldw + k0] : W[(size_t)k0 * ldw + n];
      if (k1 < K) f1 = transB ? W[(size_t)n * ldw + k1] : W[(size_t)k1 * ldw + n];
    }
    dst[d] = (unsigned)f2bf(f0) | ((unsigned)f2bf(f1) << 16);
  }
}

// ------------------------------------------------- block-tiled WMMA GEMM (128x128 per block)
// 8 waves; wave w: msubs {2*(w&3), 2*(w&3)+1}, nsubs {4*(w>>2) .. +3}; 8 accumulators.
// A packed kt-major, B packed kt-major. Double-buffered async global->LDS staging.
// Requires: M == Mt*16 with Mt % 8 == 0, Nt % 8 == 0, grid = (Mt/8)*(Nt/8).
__global__ void __launch_bounds__(256)
wmma_gemm_lds_kernel(const unsigned* __restrict__ pA, const unsigned* __restrict__ pB,
                     const float* __restrict__ bias, float* __restrict__ C,
                     int Mt, int Nt, int Kt, int N, int ldc) {
  __shared__ unsigned ldsA[2][2048];   // 8 tiles * 256 dwords
  __shared__ unsigned ldsB[2][2048];

  const int nBlocks = Nt >> 3;
  const int bm = blockIdx.x / nBlocks;
  const int bn = blockIdx.x % nBlocks;
  const int tid  = threadIdx.x;
  const int lane = tid & 31;
  const int w    = tid >> 5;
  const int wm   = w & 3;    // 4 M-groups of 32 rows
  const int wn   = w >> 2;   // 2 N-groups of 64 cols

  const unsigned* srcA = pA + ((size_t)bm * 8) * 256;   // + kt*Mt*256
  const unsigned* srcB = pB + ((size_t)bn * 8) * 256;   // + kt*Nt*256
  const size_t aKStep = (size_t)Mt * 256;
  const size_t bKStep = (size_t)Nt * 256;

  auto issue = [&](int kt, int buf) {
    const unsigned* ga = srcA + (size_t)kt * aKStep + tid * 4;
    const unsigned* gb = srcB + (size_t)kt * bKStep + tid * 4;
    async_cp16(ga,        &ldsA[buf][tid * 4]);
    async_cp16(ga + 1024, &ldsA[buf][tid * 4 + 1024]);
    async_cp16(gb,        &ldsB[buf][tid * 4]);
    async_cp16(gb + 1024, &ldsB[buf][tid * 4 + 1024]);
  };

  union Frag { uint4 q[2]; v16bf v; };
  v8f acc[2][4] = {};

  issue(0, 0);
  for (int kt = 0; kt < Kt; ++kt) {
    const int buf = kt & 1;
    __builtin_amdgcn_s_wait_asynccnt(0);
    __syncthreads();                       // stage[buf] visible to all waves
    if (kt + 1 < Kt) issue(kt + 1, buf ^ 1);

    Frag a[2], b[4];
#pragma unroll
    for (int i = 0; i < 2; ++i) {
      const unsigned* p = &ldsA[buf][(wm * 2 + i) * 256 + lane * 8];
      a[i].q[0] = *(const uint4*)p;
      a[i].q[1] = *(const uint4*)(p + 4);
    }
#pragma unroll
    for (int j = 0; j < 4; ++j) {
      const unsigned* p = &ldsB[buf][(wn * 4 + j) * 256 + lane * 8];
      b[j].q[0] = *(const uint4*)p;
      b[j].q[1] = *(const uint4*)(p + 4);
    }
#pragma unroll
    for (int i = 0; i < 2; ++i)
#pragma unroll
      for (int j = 0; j < 4; ++j)
        acc[i][j] = __builtin_amdgcn_wmma_f32_16x16x32_bf16(
            false, a[i].v, false, b[j].v, (short)0, acc[i][j], false, false);
    __syncthreads();                       // all reads of stage[buf] done
  }

  // store: m = bm*128 + msub*16 + hi*8 + r ; n = bn*128 + nsub*16 + (lane&15)
#pragma unroll
  for (int i = 0; i < 2; ++i) {
#pragma unroll
    for (int j = 0; j < 4; ++j) {
      int m0 = bm * 128 + (wm * 2 + i) * 16 + (lane >> 4) * 8;
      int n  = bn * 128 + (wn * 4 + j) * 16 + (lane & 15);
      if (n < N) {
        float bv = bias ? bias[n] : 0.f;
#pragma unroll
        for (int r = 0; r < 8; ++r)
          C[(size_t)(m0 + r) * ldc + n] = acc[i][j][r] + bv;
      }
    }
  }
}

// ------------------------------------------------- simple wave-per-tile WMMA GEMM (small GEMMs)
__global__ void __launch_bounds__(256)
wmma_gemm_kernel(const unsigned* __restrict__ pA, const unsigned* __restrict__ pB,
                 const float* __restrict__ bias, float* __restrict__ C,
                 int Mt, int Nt, int Kt, int M, int N, int ldc) {
  int wid  = (int)((blockIdx.x * blockDim.x + threadIdx.x) >> 5);
  int lane = (int)(threadIdx.x & 31);
  if (wid >= Mt * Nt) return;            // wave-uniform: EXEC stays all-ones for WMMA
  int nt = wid % Nt;
  int mt = wid / Nt;

  const uint4* a4 = (const uint4*)(pA + (((size_t)mt * Kt) * 32 + lane) * 8);
  const uint4* b4 = (const uint4*)(pB + (((size_t)nt) * 32 + lane) * 8);
  const size_t bStep = (size_t)Nt * 64;  // uint4 units (256 dwords per tile)

  union Frag { uint4 q[2]; v16bf v; };
  v8f acc = {};
  int kt = 0;
  for (; kt + 2 <= Kt; kt += 2) {
    Frag a0, b0, a1, b1;
    a0.q[0] = a4[0];     a0.q[1] = a4[1];
    b0.q[0] = b4[0];     b0.q[1] = b4[1];
    a1.q[0] = a4[64];    a1.q[1] = a4[65];
    b1.q[0] = b4[bStep]; b1.q[1] = b4[bStep + 1];
    acc = __builtin_amdgcn_wmma_f32_16x16x32_bf16(false, a0.v, false, b0.v,
                                                  (short)0, acc, false, false);
    acc = __builtin_amdgcn_wmma_f32_16x16x32_bf16(false, a1.v, false, b1.v,
                                                  (short)0, acc, false, false);
    a4 += 128;
    b4 += 2 * bStep;
  }
  if (kt < Kt) {
    Frag a0, b0;
    a0.q[0] = a4[0]; a0.q[1] = a4[1];
    b0.q[0] = b4[0]; b0.q[1] = b4[1];
    acc = __builtin_amdgcn_wmma_f32_16x16x32_bf16(false, a0.v, false, b0.v,
                                                  (short)0, acc, false, false);
  }

  int m0 = mt * 16 + (lane >> 4) * 8;
  int n  = nt * 16 + (lane & 15);
  if (n < N) {
    float bv = bias ? bias[n] : 0.f;
#pragma unroll
    for (int r = 0; r < 8; ++r) {
      int m = m0 + r;
      if (m < M) C[(size_t)m * ldc + n] = acc[r] + bv;
    }
  }
}

// ------------------------------------------------- misc elementwise
__global__ void vec_add_kernel(const float* a, const float* b, float* o, int n) {
  int i = blockIdx.x * blockDim.x + threadIdx.x;
  if (i < n) o[i] = a[i] + b[i];
}
__global__ void fill_zero_kernel(float* p, int n) {
  int i = blockIdx.x * blockDim.x + threadIdx.x;
  if (i < n) p[i] = 0.f;
}
// dst[c*R + r] = src[r*C + c]   (src is [R,C])
__global__ void transpose_kernel(const float* __restrict__ src, float* __restrict__ dst,
                                 int R, int C) {
  int i = blockIdx.x * blockDim.x + threadIdx.x;
  if (i >= R * C) return;
  int r = i / C, c = i % C;
  dst[(size_t)c * R + r] = src[i];
}

// ------------------------------------------------- LSTM recurrence (one persistent block)
__global__ void __launch_bounds__(1024)
lstm_rec_kernel(const float* __restrict__ xW,    // [T, 4H] = x@Wih^T + (bih+bhh)
                const float* __restrict__ WhhT,  // [H, 4H]
                float* __restrict__ hseq,        // [T, H]
                int T) {
  __shared__ float sh_h[H_DIM];
  __shared__ float sh_c[H_DIM];
  __shared__ float sh_g[H4];
  int tid = threadIdx.x;
  if (tid < H_DIM) { sh_h[tid] = 0.f; sh_c[tid] = 0.f; }
  __syncthreads();
  const int g0 = tid, g1 = tid + 1024, g2 = tid + 2048;
  const bool has2 = (g2 < H4);
  for (int t = 0; t < T; ++t) {
    const float* xwt = xW + (size_t)t * H4;
    float acc0 = xwt[g0];
    float acc1 = xwt[g1];
    float acc2 = has2 ? xwt[g2] : 0.f;
    for (int k = 0; k < H_DIM; ++k) {
      float hv = sh_h[k];
      const float* row = WhhT + (size_t)k * H4;
      if (k + 8 < H_DIM) __builtin_prefetch(WhhT + (size_t)(k + 8) * H4 + g0, 0, 1);
      acc0 = fmaf(hv, row[g0], acc0);
      acc1 = fmaf(hv, row[g1], acc1);
      acc2 = fmaf(hv, has2 ? row[g2] : 0.f, acc2);
    }
    sh_g[g0] = acc0;
    sh_g[g1] = acc1;
    if (has2) sh_g[g2] = acc2;
    __syncthreads();
    if (tid < H_DIM) {
      float ig = 1.f / (1.f + expf(-sh_g[tid]));
      float fg = 1.f / (1.f + expf(-sh_g[H_DIM + tid]));
      float gg = tanhf(sh_g[2 * H_DIM + tid]);
      float og = 1.f / (1.f + expf(-sh_g[3 * H_DIM + tid]));
      float c  = fg * sh_c[tid] + ig * gg;
      float h  = og * tanhf(c);
      sh_c[tid] = c;
      sh_h[tid] = h;
      hseq[(size_t)t * H_DIM + tid] = h;
    }
    __syncthreads();
  }
}

// ------------------------------------------------- GCN degree (with self loops)
__global__ void deg_init_kernel(float* deg) {
  int i = blockIdx.x * blockDim.x + threadIdx.x;
  if (i < NNODES) deg[i] = 1.f;
}
__global__ void deg_accum_kernel(const int* __restrict__ ei, float* deg) {
  int e = blockIdx.x * blockDim.x + threadIdx.x;
  if (e < NEDGES) atomicAdd(&deg[ei[NEDGES + e]], 1.f);   // row 1 = dst
}

// ------------------------------------------------- GCN scatter: agg[dst] += norm * h[src]
__global__ void gcn_scatter_kernel(const int* __restrict__ ei, const float* __restrict__ deg,
                                   const float* __restrict__ h, float* __restrict__ agg, int F) {
  int e = blockIdx.x;
  int src, dst;
  if (e < NEDGES) { src = ei[e]; dst = ei[NEDGES + e]; }
  else            { src = dst = e - NEDGES; }             // self loop
  float norm = rsqrtf(deg[src]) * rsqrtf(deg[dst]);
  for (int f = threadIdx.x; f < F; f += blockDim.x)
    atomicAdd(&agg[(size_t)dst * F + f], norm * h[(size_t)src * F + f]);
}

__global__ void bias_leaky_kernel(const float* __restrict__ agg, const float* __restrict__ b,
                                  float* __restrict__ y, int F) {
  int i = blockIdx.x * blockDim.x + threadIdx.x;
  if (i >= NNODES * F) return;
  float v = agg[i] + b[i % F];
  y[i] = (v > 0.f) ? v : 0.01f * v;
}

// ------------------------------------------------- batch-norm (training, biased var)
__global__ void bn_stats_kernel(const float* __restrict__ y, float* mean, float* istd, int F) {
  __shared__ float rs[256], rq[256];
  int f = blockIdx.x;
  float s = 0.f, q = 0.f;
  for (int r = threadIdx.x; r < NNODES; r += blockDim.x) {
    float v = y[(size_t)r * F + f];
    s += v; q += v * v;
  }
  rs[threadIdx.x] = s; rq[threadIdx.x] = q;
  __syncthreads();
  for (int st = 128; st > 0; st >>= 1) {
    if ((int)threadIdx.x < st) { rs[threadIdx.x] += rs[threadIdx.x + st];
                                 rq[threadIdx.x] += rq[threadIdx.x + st]; }
    __syncthreads();
  }
  if (threadIdx.x == 0) {
    float m = rs[0] / (float)NNODES;
    float var = rq[0] / (float)NNODES - m * m;
    mean[f] = m;
    istd[f] = rsqrtf(var + 1e-5f);
  }
}
__global__ void bn_apply_kernel(const float* __restrict__ y, const float* __restrict__ mean,
                                const float* __restrict__ istd, float* __restrict__ x, int F) {
  int i = blockIdx.x * blockDim.x + threadIdx.x;
  if (i >= NNODES * F) return;
  int f = i % F;
  x[i] = (y[i] - mean[f]) * istd[f];
}

// ------------------------------------------------- segment-sum + concat [16,52]
__global__ void seg_concat_kernel(const float* __restrict__ x,      // [1024, 50]
                                  const float* __restrict__ gender, // [16]
                                  const float* __restrict__ handed, // [16]
                                  float* __restrict__ xcat) {       // [16, 52]
  int b = blockIdx.x;
  int f = threadIdx.x;
  if (f < 50) {
    float s = 0.f;
    for (int r = 0; r < NNODES / BSZ; ++r) s += x[(size_t)(b * (NNODES / BSZ) + r) * 50 + f];
    xcat[b * 52 + f] = s;
  } else if (f == 50) xcat[b * 52 + 50] = gender[b];
  else if (f == 51)   xcat[b * 52 + 51] = handed[b];
}

// ------------------------------------------------- MLP head (single block)
__global__ void fcn_head_kernel(const float* __restrict__ xcat,
                                const float* __restrict__ W1, const float* __restrict__ b1,
                                const float* __restrict__ W2, const float* __restrict__ b2,
                                const float* __restrict__ W3, const float* __restrict__ b3,
                                float* __restrict__ out) {
  __shared__ float t1[BSZ * 32];
  __shared__ float t2[BSZ * 16];
  int tid = threadIdx.x;
  for (int idx = tid; idx < BSZ * 32; idx += blockDim.x) {
    int b = idx / 32, o = idx % 32;
    float s = b1[o];
    for (int i = 0; i < 52; ++i) s += xcat[b * 52 + i] * W1[o * 52 + i];
    t1[idx] = s;
  }
  __syncthreads();
  for (int idx = tid; idx < BSZ * 16; idx += blockDim.x) {
    int b = idx / 16, o = idx % 16;
    float s = b2[o];
    for (int i = 0; i < 32; ++i) s += t1[b * 32 + i] * W2[o * 32 + i];
    t2[idx] = s;
  }
  __syncthreads();
  for (int b = tid; b < BSZ; b += blockDim.x) {
    float s = b3[0];
    for (int i = 0; i < 16; ++i) s += t2[b * 16 + i] * W3[i];
    out[b] = s;
  }
}

// ================================================================= host
extern "C" void kernel_launch(void* const* d_in, const int* in_sizes, int n_in,
                              void* d_out, int out_size, void* d_ws, size_t ws_size,
                              hipStream_t stream) {
  (void)in_sizes; (void)n_in; (void)out_size; (void)ws_size;

  const float* x_in   = (const float*)d_in[0];
  const int*   ei     = (const int*)d_in[1];
  const float* gender = (const float*)d_in[2];
  const float* handed = (const float*)d_in[3];
  const float* Wih[3] = { (const float*)d_in[4],  (const float*)d_in[8],  (const float*)d_in[12] };
  const float* Whh[3] = { (const float*)d_in[5],  (const float*)d_in[9],  (const float*)d_in[13] };
  const float* bih[3] = { (const float*)d_in[6],  (const float*)d_in[10], (const float*)d_in[14] };
  const float* bhh[3] = { (const float*)d_in[7],  (const float*)d_in[11], (const float*)d_in[15] };
  const float* gW[4]  = { (const float*)d_in[16], (const float*)d_in[18],
                          (const float*)d_in[20], (const float*)d_in[22] };
  const float* gB[4]  = { (const float*)d_in[17], (const float*)d_in[19],
                          (const float*)d_in[21], (const float*)d_in[23] };
  const float* fW1 = (const float*)d_in[24]; const float* fb1 = (const float*)d_in[25];
  const float* fW2 = (const float*)d_in[26]; const float* fb2 = (const float*)d_in[27];
  const float* fW3 = (const float*)d_in[28]; const float* fb3 = (const float*)d_in[29];

  const int Mt  = NNODES / 16;                 // 64
  const int Kt0 = (LEN_IN + 31) / 32;          // 266
  const int KtH = H_DIM / 32;                  // 20
  const int NtL = H4 / 16;                     // 160

  // ---- bump allocator over d_ws
  char*  ws  = (char*)d_ws;
  size_t off = 0;
  auto alloc = [&](size_t bytes) -> char* {
    char* p = ws + off;
    off = (off + bytes + 255) & ~(size_t)255;
    return p;
  };
  unsigned* pAx   = (unsigned*)alloc((size_t)Mt * Kt0 * 256 * 4);   // packed x_in (kt-major)
  unsigned* pB    = (unsigned*)alloc((size_t)Kt0 * NtL * 256 * 4);  // packed Wih (reused per layer)
  unsigned* pAh   = (unsigned*)alloc((size_t)Mt * KtH * 256 * 4);   // packed hidden-seq A
  unsigned* pBg   = (unsigned*)alloc((size_t)KtH * KtH * 256 * 4);  // packed GCN weights
  float* biasc    = (float*)alloc((size_t)H4 * 4);
  float* xW       = (float*)alloc((size_t)NNODES * H4 * 4);
  float* WhhT     = (float*)alloc((size_t)H_DIM * H4 * 4);
  float* hA       = (float*)alloc((size_t)NNODES * H_DIM * 4);
  float* hB       = (float*)alloc((size_t)NNODES * H_DIM * 4);
  float* gH       = (float*)alloc((size_t)NNODES * 320 * 4);
  float* gAgg     = (float*)alloc((size_t)NNODES * 320 * 4);
  float* gY       = (float*)alloc((size_t)NNODES * 320 * 4);
  float* xA       = (float*)alloc((size_t)NNODES * 320 * 4);
  float* xB       = (float*)alloc((size_t)NNODES * 320 * 4);
  float* deg      = (float*)alloc((size_t)NNODES * 4);
  float* meanb    = (float*)alloc(320 * 4);
  float* istdb    = (float*)alloc(320 * 4);
  float* xcat     = (float*)alloc((size_t)BSZ * 52 * 4);

  auto cdiv = [](int a, int b) { return (a + b - 1) / b; };

  // ================= LSTM =================
  float* hseq[2] = { hA, hB };
  for (int l = 0; l < 3; ++l) {
    const int K  = (l == 0) ? LEN_IN : H_DIM;
    const int Kt = (l == 0) ? Kt0 : KtH;
    unsigned* pA = (l == 0) ? pAx : pAh;
    const float* src = (l == 0) ? x_in : hseq[(l + 1) & 1];

    // pack A kt-major for the block-tiled async kernel
    pack_a_kernel<<<cdiv(Mt * Kt * 32, 256), 256, 0, stream>>>(src, pA, NNODES, K, K, Mt, Kt, 1);
    vec_add_kernel<<<cdiv(H4, 256), 256, 0, stream>>>(bih[l], bhh[l], biasc, H4);
    pack_b_kernel<<<cdiv(Kt * NtL * 32, 256), 256, 0, stream>>>(Wih[l], pB, K, H4, Kt, NtL, K, 1);
    // xW = src @ Wih^T + bias : [1024, 2560]; 128x128 block tiles with async LDS staging
    wmma_gemm_lds_kernel<<<(Mt / 8) * (NtL / 8), 256, 0, stream>>>(
        pA, pB, biasc, xW, Mt, NtL, Kt, H4, H4);
    transpose_kernel<<<cdiv(H4 * H_DIM, 256), 256, 0, stream>>>(Whh[l], WhhT, H4, H_DIM);
    lstm_rec_kernel<<<1, 1024, 0, stream>>>(xW, WhhT, hseq[l & 1], NNODES);
  }
  float* xcur = hseq[0];   // layer 2 wrote hseq[2&1] = hA

  // ================= GCN degrees =================
  deg_init_kernel<<<cdiv(NNODES, 256), 256, 0, stream>>>(deg);
  deg_accum_kernel<<<cdiv(NEDGES, 256), 256, 0, stream>>>(ei, deg);

  // ================= GCN layers =================
  const int gin[4]  = { 640, 320, 180, 90 };
  const int gout[4] = { 320, 180, 90, 50 };
  float* gx[2] = { xA, xB };
  for (int l = 0; l < 4; ++l) {
    const int Kin = gin[l], Nout = gout[l];
    const int Kt = (Kin + 31) / 32;
    const int Nt = (Nout + 15) / 16;
    const float* src = (l == 0) ? xcur : gx[(l + 1) & 1];

    pack_a_kernel<<<cdiv(Mt * Kt * 32, 256), 256, 0, stream>>>(src, pAh, NNODES, Kin, Kin, Mt, Kt, 0);
    pack_b_kernel<<<cdiv(Kt * Nt * 32, 256), 256, 0, stream>>>(gW[l], pBg, Kin, Nout, Kt, Nt, Nout, 0);
    wmma_gemm_kernel<<<cdiv(Mt * Nt * 32, 256), 256, 0, stream>>>(
        pAh, pBg, nullptr, gH, Mt, Nt, Kt, NNODES, Nout, Nout);

    fill_zero_kernel<<<cdiv(NNODES * Nout, 256), 256, 0, stream>>>(gAgg, NNODES * Nout);
    gcn_scatter_kernel<<<NEDGES + NNODES, 128, 0, stream>>>(ei, deg, gH, gAgg, Nout);
    bias_leaky_kernel<<<cdiv(NNODES * Nout, 256), 256, 0, stream>>>(gAgg, gB[l], gY, Nout);
    bn_stats_kernel<<<Nout, 256, 0, stream>>>(gY, meanb, istdb, Nout);
    bn_apply_kernel<<<cdiv(NNODES * Nout, 256), 256, 0, stream>>>(gY, meanb, istdb, gx[l & 1], Nout);
  }
  float* xg = gx[3 & 1];   // final GCN output [1024, 50]

  // ================= head =================
  seg_concat_kernel<<<BSZ, 64, 0, stream>>>(xg, gender, handed, xcat);
  fcn_head_kernel<<<1, 256, 0, stream>>>(xcat, fW1, fb1, fW2, fb2, fW3, fb3, (float*)d_out);
}
